// DecoderLayer_67912022884385
// MI455X (gfx1250) — compile-verified, matrix operand order checked by reading, not run on hardware
//
#include <hip/hip_runtime.h>
#include <hip/hip_bf16.h>

// Problem constants (match reference)
#define B_ 4
#define T_ 2048
#define S_ 2048
#define E_ 512
#define H_ 8
#define D_ 64
#define F_ (4 * E_)   // FFN hidden width = 2048
#define NBT (B_ * T_) // 8192 flattened query rows
#define NBS (B_ * S_) // 8192 flattened kv rows

typedef __bf16 bf16;
typedef __bf16 v16bf __attribute__((ext_vector_type(16)));
typedef float  v8f   __attribute__((ext_vector_type(8)));
typedef unsigned int v4u __attribute__((ext_vector_type(4)));

union FragBF {
    v16bf        v;
    unsigned int u[8];
    bf16         h[16];
};

__device__ __forceinline__ v8f zero8() {
    v8f z = {0.f, 0.f, 0.f, 0.f, 0.f, 0.f, 0.f, 0.f};
    return z;
}

__device__ __forceinline__ unsigned short bf16_bits(float f) {
    bf16 h = (bf16)f;
    return __builtin_bit_cast(unsigned short, h);
}

// Load a 16x32 bf16 A-fragment (or symmetric B-fragment from a [N,K] row-major
// "B-transposed" matrix). Per ISA: lane (0..15) holds row M=lane, halves j map
// to K = 16*(j>=8) + 8*(lane>=16) + (j&7). Pairs (2j,2j+1) are K-consecutive ->
// aligned dword loads (compiler merges into 2x global_load_b128).
__device__ __forceinline__ void load_fragA(FragBF& f, const bf16* __restrict__ base,
                                           int ld, int row, int k0, int laneHalf) {
    const unsigned char* p = (const unsigned char*)(base + (long)row * ld + k0 + laneHalf);
#pragma unroll
    for (int j = 0; j < 8; ++j) {
        int k = ((j >> 2) << 4) + ((j & 3) << 1);
        f.u[j] = *(const unsigned int*)(p + 2 * k);
    }
}

__device__ __forceinline__ v8f wmma_bf16(const FragBF& a, const FragBF& b, v8f c) {
    return __builtin_amdgcn_wmma_f32_16x16x32_bf16(false, a.v, false, b.v,
                                                   (short)0, c, false, false);
}

// ---------------------------------------------------------------------------
// Generic WMMA GEMM: C[M,N] = A[M,K](bf16,row-major) * BT[N,K](bf16,row-major)^T
// Each wave computes a 32x64 tile (2 A-frags + 4 B-frags -> 8 WMMAs per k-step,
// 0.75 fragment loads per WMMA).
// MODE 0: store bf16                      (projections)
// MODE 1: store bf16, +bias, ReLU         (FFN hidden)
// MODE 2: store f32,  +bias, +residual    (output proj / FFN out)
// MODE 3: store bf16 TRANSPOSED C^T[N,M]  (V projection -> V^T for attention;
//         each lane's 8 rows are contiguous -> one b128 store per half-tile)
// Batched over grid.z:  A += (z/zDivA)*sA ; BT += (z%zModB)*sB ; C += z*sC
// ---------------------------------------------------------------------------
template <int MODE>
__global__ __launch_bounds__(256) void gemm_wmma(
    const bf16* __restrict__ A, const bf16* __restrict__ BT,
    const float* __restrict__ bias, const float* resid, void* Cout,
    int M, int N, int K, int lda, int ldb, int ldc,
    int zDivA, long sA, int zModB, long sB, long sC) {
    const int lane     = threadIdx.x & 31;
    const int ln16     = lane & 15;
    const int hi       = lane >> 4;
    const int laneHalf = hi << 3;

    int wave     = blockIdx.x * (blockDim.x >> 5) + (threadIdx.x >> 5);
    int rowTiles = M >> 5;   // 32-row wave tiles
    int colSup   = N >> 6;   // 64-col wave tiles
    if (wave >= rowTiles * colSup) return;
    int rowBase = (wave % rowTiles) << 5;
    int colBase = (wave / rowTiles) << 6;

    int z = blockIdx.z;
    A += (long)(z / zDivA) * sA;
    BT += (long)(z % zModB) * sB;
    long cOff = (long)z * sC;

    v8f acc[2][4];
#pragma unroll
    for (int h2 = 0; h2 < 2; ++h2)
#pragma unroll
        for (int nt = 0; nt < 4; ++nt) acc[h2][nt] = zero8();

    FragBF fa0, fa1, fb;
    for (int k0 = 0; k0 < K; k0 += 32) {
        load_fragA(fa0, A, lda, rowBase + ln16, k0, laneHalf);
        load_fragA(fa1, A, lda, rowBase + 16 + ln16, k0, laneHalf);
#pragma unroll
        for (int nt = 0; nt < 4; ++nt) {
            load_fragA(fb, BT, ldb, colBase + nt * 16 + ln16, k0, laneHalf);
            acc[0][nt] = wmma_bf16(fa0, fb, acc[0][nt]);
            acc[1][nt] = wmma_bf16(fa1, fb, acc[1][nt]);
        }
    }

#pragma unroll
    for (int h2 = 0; h2 < 2; ++h2) {
#pragma unroll
        for (int nt = 0; nt < 4; ++nt) {
            int c = colBase + nt * 16 + ln16;
            if (MODE == 3) {
                // transposed store: rows rowBase+h2*16+8*hi+0..7 contiguous
                v4u pk;
#pragma unroll
                for (int j = 0; j < 4; ++j) {
                    unsigned lo = bf16_bits(acc[h2][nt][2 * j]);
                    unsigned hh = bf16_bits(acc[h2][nt][2 * j + 1]);
                    pk[j] = lo | (hh << 16);
                }
                long r0 = rowBase + h2 * 16 + (hi << 3);
                *(v4u*)((bf16*)Cout + cOff + (long)c * ldc + r0) = pk;
            } else {
#pragma unroll
                for (int i = 0; i < 8; ++i) {
                    int  r   = rowBase + h2 * 16 + i + (hi << 3);
                    long idx = cOff + (long)r * ldc + c;
                    float vv = acc[h2][nt][i];
                    if (MODE == 0) {
                        ((bf16*)Cout)[idx] = (bf16)vv;
                    } else if (MODE == 1) {
                        vv += bias[c];
                        vv = vv > 0.f ? vv : 0.f;
                        ((bf16*)Cout)[idx] = (bf16)vv;
                    } else {
                        vv += bias[c] + resid[idx];
                        ((float*)Cout)[idx] = vv;
                    }
                }
            }
        }
    }
}

// ---------------------------------------------------------------------------
// Flash-style attention. One wave = 16 query rows, full D=64 output in regs.
// Streams S in 32-wide chunks: 2 WMMAs (QK^T, K-dim=64), online softmax with
// shfl reductions, P staged through per-wave LDS tile, 4 WMMAs (P*V^T-frag).
// Qh/Kh: [B*H, T|S, 64] bf16.  VhT: [B*H, 64, S] bf16 (pre-transposed).
// Out: concat layout [B*T, H*64] bf16.
// ---------------------------------------------------------------------------
template <int CAUSAL>
__global__ __launch_bounds__(256) void attn_wmma(
    const bf16* __restrict__ Qh, const bf16* __restrict__ Kh,
    const bf16* __restrict__ VhT, bf16* __restrict__ Out, int T, int S) {
    __shared__ bf16 pbuf[8 * 16 * 32];

    const int lane     = threadIdx.x & 31;
    const int ln16     = lane & 15;
    const int hi       = lane >> 4;
    const int laneHalf = hi << 3;
    const int wInB     = threadIdx.x >> 5;

    int z = blockIdx.z;  // b*H + h
    int b = z / H_, h = z % H_;
    const bf16* Q   = Qh + (long)z * T * D_;
    const bf16* Kb  = Kh + (long)z * S * D_;
    const bf16* VbT = VhT + (long)z * D_ * S;

    int qTile = blockIdx.x * (blockDim.x >> 5) + wInB;
    if (qTile >= (T >> 4)) return;
    int   qBase = qTile << 4;
    bf16* pw    = pbuf + wInB * (16 * 32);

    FragBF fq0, fq1;
    load_fragA(fq0, Q, D_, qBase + ln16, 0, laneHalf);
    load_fragA(fq1, Q, D_, qBase + ln16, 32, laneHalf);

    v8f acc[4];
#pragma unroll
    for (int nt = 0; nt < 4; ++nt) acc[nt] = zero8();
    float rowMax[8], rowSum[8];
#pragma unroll
    for (int i = 0; i < 8; ++i) { rowMax[i] = -1e30f; rowSum[i] = 0.f; }

    int sEnd = CAUSAL ? (qBase + 16) : S;
    for (int s0 = 0; s0 < sEnd; s0 += 32) {
        float sc[2][8];
#pragma unroll
        for (int half = 0; half < 2; ++half) {
            int sBase = s0 + (half << 4);
            v8f sa    = zero8();
            FragBF fk;
            load_fragA(fk, Kb, D_, sBase + ln16, 0, laneHalf);
            sa = wmma_bf16(fq0, fk, sa);
            load_fragA(fk, Kb, D_, sBase + ln16, 32, laneHalf);
            sa = wmma_bf16(fq1, fk, sa);
            int s = sBase + ln16;
#pragma unroll
            for (int i = 0; i < 8; ++i) {
                float vv = sa[i] * 0.125f;  // 1/sqrt(D)
                if (CAUSAL) {
                    int t = qBase + i + (hi << 3);
                    if (s > t) vv = -1e30f;
                }
                sc[half][i] = vv;
            }
        }
        // online softmax update (rows live in 16-lane groups)
#pragma unroll
        for (int i = 0; i < 8; ++i) {
            float cm = fmaxf(sc[0][i], sc[1][i]);
#pragma unroll
            for (int m = 1; m < 16; m <<= 1) cm = fmaxf(cm, __shfl_xor(cm, m, 32));
            float nm   = fmaxf(rowMax[i], cm);
            float corr = __expf(rowMax[i] - nm);
            float p0   = __expf(sc[0][i] - nm);
            float p1   = __expf(sc[1][i] - nm);
            float ps   = p0 + p1;
#pragma unroll
            for (int m = 1; m < 16; m <<= 1) ps += __shfl_xor(ps, m, 32);
            rowSum[i] = rowSum[i] * corr + ps;
            rowMax[i] = nm;
#pragma unroll
            for (int nt = 0; nt < 4; ++nt) acc[nt][i] *= corr;
            int r = i + (hi << 3);
            pw[r * 32 + ln16]      = (bf16)p0;
            pw[r * 32 + 16 + ln16] = (bf16)p1;
        }
        __builtin_amdgcn_wave_barrier();
        asm volatile("s_wait_dscnt 0x0" ::: "memory");
        FragBF fp;
        load_fragA(fp, pw, 32, ln16, 0, laneHalf);  // C-layout -> A-frag via LDS
#pragma unroll
        for (int nt = 0; nt < 4; ++nt) {
            FragBF fv;  // B-frag from V^T [D,S]: dword loads, K=s contiguous
            load_fragA(fv, VbT, S, nt * 16 + ln16, s0, laneHalf);
            acc[nt] = wmma_bf16(fp, fv, acc[nt]);
        }
        __builtin_amdgcn_wave_barrier();
    }

    bf16* Ob = Out + (long)b * T * (H_ * D_) + h * D_;
#pragma unroll
    for (int nt = 0; nt < 4; ++nt) {
#pragma unroll
        for (int i = 0; i < 8; ++i) {
            int   t  = qBase + i + (hi << 3);
            float vv = acc[nt][i] / rowSum[i];
            Ob[(long)t * (H_ * D_) + nt * 16 + ln16] = (bf16)vv;
        }
    }
}

// ---------------------------------------------------------------------------
// LayerNorm: one block per row of E=512, outputs bf16.
// ---------------------------------------------------------------------------
__global__ __launch_bounds__(256) void ln_bf16(const float* __restrict__ x,
                                               const float* __restrict__ g,
                                               const float* __restrict__ be,
                                               bf16* __restrict__ out) {
    __shared__ float s1[256], s2[256];
    int          tid = threadIdx.x;
    long         row = blockIdx.x;
    const float* xr  = x + row * E_;
    float a  = xr[tid];
    float b2 = xr[tid + 256];
    s1[tid] = a + b2;
    s2[tid] = a * a + b2 * b2;
    __syncthreads();
    for (int st = 128; st > 0; st >>= 1) {
        if (tid < st) { s1[tid] += s1[tid + st]; s2[tid] += s2[tid + st]; }
        __syncthreads();
    }
    float mean = s1[0] * (1.0f / E_);
    float var  = s2[0] * (1.0f / E_) - mean * mean;
    float rs   = rsqrtf(var + 1e-5f);
    out[row * E_ + tid]       = (bf16)((a - mean) * rs * g[tid] + be[tid]);
    out[row * E_ + tid + 256] = (bf16)((b2 - mean) * rs * g[tid + 256] + be[tid + 256]);
}

// fp32 [nmat,K,N] -> bf16 [nmat,N,K] (weights into fragment-friendly layout)
__global__ void transpose_w(const float* __restrict__ W, bf16* __restrict__ WT,
                            int K, int N, long total) {
    long idx = (long)blockIdx.x * blockDim.x + threadIdx.x;
    if (idx >= total) return;
    long mat = idx / ((long)K * N);
    long rem = idx - mat * (long)K * N;
    int  k = (int)(rem / N), n = (int)(rem % N);
    WT[mat * (long)K * N + (long)n * K + k] = (bf16)W[idx];
}

__global__ void cvt_bf16_k(const float* __restrict__ in, bf16* __restrict__ out, long n) {
    long i = (long)blockIdx.x * blockDim.x + threadIdx.x;
    if (i < n) out[i] = (bf16)in[i];
}

__global__ void copy_f32_k(const float* __restrict__ in, float* __restrict__ out, long n) {
    long i = (long)blockIdx.x * blockDim.x + threadIdx.x;
    if (i < n) out[i] = in[i];
}

// ---------------------------------------------------------------------------
extern "C" void kernel_launch(void* const* d_in, const int* in_sizes, int n_in,
                              void* d_out, int out_size, void* d_ws, size_t ws_size,
                              hipStream_t stream) {
    (void)in_sizes; (void)n_in; (void)out_size; (void)ws_size;

    const float* q    = (const float*)d_in[0];
    const float* k    = (const float*)d_in[1];
    const float* v    = (const float*)d_in[2];
    const float* Wq_s = (const float*)d_in[3];
    const float* Wk_s = (const float*)d_in[4];
    const float* Wv_s = (const float*)d_in[5];
    const float* Wp_s = (const float*)d_in[6];
    const float* bp_s = (const float*)d_in[7];
    const float* Wq_c = (const float*)d_in[8];
    const float* Wk_c = (const float*)d_in[9];
    const float* Wv_c = (const float*)d_in[10];
    const float* Wp_c = (const float*)d_in[11];
    const float* bp_c = (const float*)d_in[12];
    const float* W1   = (const float*)d_in[13];
    const float* b1   = (const float*)d_in[14];
    const float* W2   = (const float*)d_in[15];
    const float* b2   = (const float*)d_in[16];
    const float* g1   = (const float*)d_in[17];
    const float* be1  = (const float*)d_in[18];
    const float* g2   = (const float*)d_in[19];
    const float* be2  = (const float*)d_in[20];
    const float* g3   = (const float*)d_in[21];
    const float* be3  = (const float*)d_in[22];
    float* out = (float*)d_out;  // fp32 residual stream lives here

    // workspace carve-out
    char* wp   = (char*)d_ws;
    auto  take = [&](size_t bytes) -> char* {
        char* r = wp;
        wp += (bytes + 255) & ~(size_t)255;
        return r;
    };
    bf16* xn     = (bf16*)take((size_t)NBT * E_ * 2);
    bf16* qh     = (bf16*)take((size_t)B_ * H_ * T_ * D_ * 2);
    bf16* kh     = (bf16*)take((size_t)B_ * H_ * S_ * D_ * 2);
    bf16* vhT    = (bf16*)take((size_t)B_ * H_ * D_ * S_ * 2);  // V^T per head
    bf16* concat = (bf16*)take((size_t)NBT * (H_ * D_) * 2);
    bf16* kbf    = (bf16*)take((size_t)NBS * E_ * 2);
    bf16* vbf    = (bf16*)take((size_t)NBS * E_ * 2);
    bf16* ffnh   = (bf16*)take((size_t)NBT * F_ * 2);
    bf16* WqTs   = (bf16*)take((size_t)H_ * E_ * D_ * 2);
    bf16* WkTs   = (bf16*)take((size_t)H_ * E_ * D_ * 2);
    bf16* WvTs   = (bf16*)take((size_t)H_ * E_ * D_ * 2);
    bf16* WqTc   = (bf16*)take((size_t)H_ * E_ * D_ * 2);
    bf16* WkTc   = (bf16*)take((size_t)H_ * E_ * D_ * 2);
    bf16* WvTc   = (bf16*)take((size_t)H_ * E_ * D_ * 2);
    bf16* WpTs   = (bf16*)take((size_t)(H_ * D_) * E_ * 2);
    bf16* WpTc   = (bf16*)take((size_t)(H_ * D_) * E_ * 2);
    bf16* W1T    = (bf16*)take((size_t)E_ * F_ * 2);
    bf16* W2T    = (bf16*)take((size_t)F_ * E_ * 2);

    auto blk1d = [](long n) { return (unsigned)((n + 255) / 256); };

    auto gemm = [&](int mode, const bf16* A, const bf16* BT, const float* bias,
                    const float* resid, void* C, int M, int N, int K, int lda,
                    int ldb, int ldc, int zC, int zDivA, long sA, int zModB,
                    long sB, long sC) {
        int  waves = (M >> 5) * (N >> 6);
        dim3 g((unsigned)((waves + 7) >> 3), 1, (unsigned)zC);
        if (mode == 0)
            gemm_wmma<0><<<g, 256, 0, stream>>>(A, BT, bias, resid, C, M, N, K,
                                                lda, ldb, ldc, zDivA, sA, zModB, sB, sC);
        else if (mode == 1)
            gemm_wmma<1><<<g, 256, 0, stream>>>(A, BT, bias, resid, C, M, N, K,
                                                lda, ldb, ldc, zDivA, sA, zModB, sB, sC);
        else if (mode == 2)
            gemm_wmma<2><<<g, 256, 0, stream>>>(A, BT, bias, resid, C, M, N, K,
                                                lda, ldb, ldc, zDivA, sA, zModB, sB, sC);
        else
            gemm_wmma<3><<<g, 256, 0, stream>>>(A, BT, bias, resid, C, M, N, K,
                                                lda, ldb, ldc, zDivA, sA, zModB, sB, sC);
    };

    // 0. residual init: out = q
    copy_f32_k<<<blk1d((long)NBT * E_), 256, 0, stream>>>(q, out, (long)NBT * E_);

    // 1. weight conversion (bf16, transposed to [N,K])
    long hw = (long)H_ * E_ * D_;
    transpose_w<<<blk1d(hw), 256, 0, stream>>>(Wq_s, WqTs, E_, D_, hw);
    transpose_w<<<blk1d(hw), 256, 0, stream>>>(Wk_s, WkTs, E_, D_, hw);
    transpose_w<<<blk1d(hw), 256, 0, stream>>>(Wv_s, WvTs, E_, D_, hw);
    transpose_w<<<blk1d(hw), 256, 0, stream>>>(Wq_c, WqTc, E_, D_, hw);
    transpose_w<<<blk1d(hw), 256, 0, stream>>>(Wk_c, WkTc, E_, D_, hw);
    transpose_w<<<blk1d(hw), 256, 0, stream>>>(Wv_c, WvTc, E_, D_, hw);
    long pw2 = (long)(H_ * D_) * E_;
    transpose_w<<<blk1d(pw2), 256, 0, stream>>>(Wp_s, WpTs, H_ * D_, E_, pw2);
    transpose_w<<<blk1d(pw2), 256, 0, stream>>>(Wp_c, WpTc, H_ * D_, E_, pw2);
    long w1n = (long)E_ * F_;
    transpose_w<<<blk1d(w1n), 256, 0, stream>>>(W1, W1T, E_, F_, w1n);
    transpose_w<<<blk1d(w1n), 256, 0, stream>>>(W2, W2T, F_, E_, w1n);

    // 2. k, v -> bf16 (cross-attention sources, un-normalized per reference)
    cvt_bf16_k<<<blk1d((long)NBS * E_), 256, 0, stream>>>(k, kbf, (long)NBS * E_);
    cvt_bf16_k<<<blk1d((long)NBS * E_), 256, 0, stream>>>(v, vbf, (long)NBS * E_);

    const long sQKV_A = (long)T_ * E_;  // per-batch activation stride
    const long sW     = (long)D_ * E_;  // per-head weight stride
    const long sHead  = (long)T_ * D_;  // per-(b,h) head-matrix stride
    dim3 aGrid((unsigned)((T_ / 16) / 8), 1, (unsigned)(B_ * H_));

    // ---- self attention ----
    ln_bf16<<<NBT, 256, 0, stream>>>(out, g1, be1, xn);
    gemm(0, xn, WqTs, nullptr, nullptr, qh, T_, D_, E_, E_, E_, D_,
         B_ * H_, H_, sQKV_A, H_, sW, sHead);
    gemm(0, xn, WkTs, nullptr, nullptr, kh, T_, D_, E_, E_, E_, D_,
         B_ * H_, H_, sQKV_A, H_, sW, sHead);
    gemm(3, xn, WvTs, nullptr, nullptr, vhT, T_, D_, E_, E_, E_, /*ldc=*/S_,
         B_ * H_, H_, sQKV_A, H_, sW, (long)D_ * S_);
    attn_wmma<1><<<aGrid, 256, 0, stream>>>(qh, kh, vhT, concat, T_, S_);
    gemm(2, concat, WpTs, bp_s, out, out, NBT, E_, H_ * D_, H_ * D_, H_ * D_, E_,
         1, 1, 0, 1, 0, 0);

    // ---- cross attention ----
    ln_bf16<<<NBT, 256, 0, stream>>>(out, g2, be2, xn);
    gemm(0, xn, WqTc, nullptr, nullptr, qh, T_, D_, E_, E_, E_, D_,
         B_ * H_, H_, sQKV_A, H_, sW, sHead);
    gemm(0, kbf, WkTc, nullptr, nullptr, kh, S_, D_, E_, E_, E_, D_,
         B_ * H_, H_, (long)S_ * E_, H_, sW, (long)S_ * D_);
    gemm(3, vbf, WvTc, nullptr, nullptr, vhT, S_, D_, E_, E_, E_, /*ldc=*/S_,
         B_ * H_, H_, (long)S_ * E_, H_, sW, (long)D_ * S_);
    attn_wmma<0><<<aGrid, 256, 0, stream>>>(qh, kh, vhT, concat, T_, S_);
    gemm(2, concat, WpTc, bp_c, out, out, NBT, E_, H_ * D_, H_ * D_, H_ * D_, E_,
         1, 1, 0, 1, 0, 0);

    // ---- FFN ----
    ln_bf16<<<NBT, 256, 0, stream>>>(out, g3, be3, xn);
    gemm(1, xn, W1T, b1, nullptr, ffnh, NBT, F_, E_, E_, E_, F_,
         1, 1, 0, 1, 0, 0);
    gemm(2, ffnh, W2T, b2, out, out, NBT, E_, F_, F_, F_, E_,
         1, 1, 0, 1, 0, 0);
}
